// AttentionLayer_89266600280579
// MI455X (gfx1250) — matrix operand order repaired and checked by reading, not compiled, not run on hardware
//
#include <hip/hip_runtime.h>

typedef _Float16 f16;
typedef __attribute__((ext_vector_type(16))) _Float16 v16h;
typedef __attribute__((ext_vector_type(8)))  _Float16 v8h;
typedef __attribute__((ext_vector_type(8)))  float    v8f;

#define MD 512      // model dim
#define KD 64       // key dim
#define S_LEN 4096  // sequence length

static __device__ __forceinline__ v8f wmma_f16(v16h a, v16h b, v8f c) {
  // D(16x16 f32) = A(16x32 f16) * B(32x16 f16) + C
  return __builtin_amdgcn_wmma_f32_16x16x32_f16(false, a, false, b, (short)0, c,
                                                false, false);
}

// Async global -> LDS copy of 16 bytes (CDNA5 GLOBAL_LOAD_ASYNC_TO_LDS_B128,
// tracked by ASYNCcnt). lds_byte_off = wave-logical LDS byte address (low 32
// bits of the generic shared pointer), gptr = 64-bit global address.
static __device__ __forceinline__ void async_b128(unsigned lds_byte_off,
                                                  const void* gptr) {
  asm volatile("global_load_async_to_lds_b128 %0, %1, off"
               :
               : "v"(lds_byte_off), "v"((unsigned long long)gptr)
               : "memory");
}

static __device__ __forceinline__ unsigned lds_off(const void* p) {
  return (unsigned)(unsigned long long)p;  // generic LDS addr: offset in [31:0]
}

// ---------------------------------------------------------------------------
// Kernel 0: convert weights to f16 and transpose to [3][KD][MD] so that a
// WMMA B-fragment (K contiguous for a fixed output column) is one contiguous
// 32-byte load per lane.
// ---------------------------------------------------------------------------
__global__ void prep_weights(const float* __restrict__ Wq,
                             const float* __restrict__ Wk,
                             const float* __restrict__ Wv,
                             f16* __restrict__ wT) {
  int idx = blockIdx.x * blockDim.x + threadIdx.x;
  const int n = 3 * MD * KD;
  if (idx >= n) return;
  int w = idx / (MD * KD);
  int r = (idx / KD) % MD;  // k index (0..511)
  int c = idx % KD;         // n index (0..63)
  const float* W = (w == 0) ? Wq : (w == 1) ? Wk : Wv;
  wT[(size_t)w * KD * MD + (size_t)c * MD + r] = (f16)W[(size_t)r * KD + c];
}

// ---------------------------------------------------------------------------
// Kernel 1: fused Q/K/V projection with ReLU.
// One wave per 16-row tile. 4 waves per block. Per k-step, all 12 weight
// B-fragments are loaded in one batch (single clause) before the 12 WMMAs so
// load latency overlaps matrix issue. Q,K stored row-major f16 [rows][64];
// V stored transposed per batch [B][64][S] (for PV B-fragments).
// ---------------------------------------------------------------------------
__global__ __launch_bounds__(128) void proj_qkv(
    const float* __restrict__ x1, const float* __restrict__ x2,
    const f16* __restrict__ wT,  // [3][KD][MD] f16
    const float* __restrict__ bq, const float* __restrict__ bk,
    const float* __restrict__ bv,
    f16* __restrict__ qws, f16* __restrict__ kws, f16* __restrict__ vtws,
    int rows) {
  const int lane = threadIdx.x & 31;
  const int wave = threadIdx.x >> 5;
  const int tile = blockIdx.x * 4 + wave;
  const int row0 = tile * 16;
  if (row0 >= rows) return;

  const int hi   = lane >> 4;   // selects K half of the A fragment
  const int n0   = lane & 15;   // A-row / B-column within tile
  const int koff = hi * 16;

  const float* x1row = x1 + (size_t)(row0 + n0) * MD;
  const float* x2row = x2 + (size_t)(row0 + n0) * MD;
  const f16* wqT = wT;
  const f16* wkT = wT + (size_t)KD * MD;
  const f16* wvT = wT + (size_t)2 * KD * MD;

  v8f aq[4], ak[4], av[4];
#pragma unroll
  for (int nt = 0; nt < 4; ++nt) { aq[nt] = 0.0f; ak[nt] = 0.0f; av[nt] = 0.0f; }

  for (int kb = 0; kb < MD; kb += 32) {
    // A fragments (16x32 f16): lane row = lane%16; elems 0..15 -> K=koff+e
    // (contiguous order, matching the B fragments).
    v16h a1, a2;
    const float* p1 = x1row + kb + koff;
    const float* p2 = x2row + kb + koff;
#pragma unroll
    for (int e = 0; e < 8; ++e) {
      a1[e] = (f16)p1[e];      a1[e + 8] = (f16)p1[e + 8];
      a2[e] = (f16)p2[e];      a2[e + 8] = (f16)p2[e + 8];
    }
    const int kfrag = kb + hi * 16;  // B frag: K = e + 16*(lane>=16)
    // Stage all 12 weight fragments first (one load clause), then issue the
    // 12 WMMAs -- loads retire underneath earlier WMMAs instead of stalling
    // each one on s_wait_loadcnt 0.
    v16h bf[12];
#pragma unroll
    for (int nt = 0; nt < 4; ++nt) {
      const int col = nt * 16 + n0;
      bf[nt]     = *(const v16h*)(wqT + (size_t)col * MD + kfrag);
      bf[4 + nt] = *(const v16h*)(wkT + (size_t)col * MD + kfrag);
      bf[8 + nt] = *(const v16h*)(wvT + (size_t)col * MD + kfrag);
    }
#pragma unroll
    for (int nt = 0; nt < 4; ++nt) {
      aq[nt] = wmma_f16(a1, bf[nt],     aq[nt]);
      ak[nt] = wmma_f16(a2, bf[4 + nt], ak[nt]);
      av[nt] = wmma_f16(a2, bf[8 + nt], av[nt]);
    }
  }

  // Epilogue: C layout -> lane holds (M = g + 8*hi, N = lane&15).
  const int b = row0 / S_LEN;
#pragma unroll
  for (int nt = 0; nt < 4; ++nt) {
    const int n = nt * 16 + n0;
    const float bqv = bq[n], bkv = bk[n], bvv = bv[n];
#pragma unroll
    for (int g = 0; g < 8; ++g) {
      const int row = row0 + g + 8 * hi;
      float qf = aq[nt][g] + bqv; qf = qf > 0.f ? qf : 0.f;
      float kf = ak[nt][g] + bkv; kf = kf > 0.f ? kf : 0.f;
      float vf = av[nt][g] + bvv; vf = vf > 0.f ? vf : 0.f;
      qws[(size_t)row * KD + n] = (f16)qf;
      kws[(size_t)row * KD + n] = (f16)kf;
      vtws[((size_t)b * KD + n) * S_LEN + (row - b * S_LEN)] = (f16)vf;
    }
  }
}

// ---------------------------------------------------------------------------
// Kernel 2: fused causal flash-attention.
// One block = 64 query rows = 4 waves x 16-row tiles sharing the same key
// range. K/V tiles (32 keys) are double-buffered in LDS, filled with async
// global->LDS loads (ASYNCcnt) and prefetched one tile ahead. Per-wave P
// staging through LDS with s_wait_dscnt (wave-local, no barrier).
// ---------------------------------------------------------------------------
__global__ __launch_bounds__(128) void attn_fused(
    const f16* __restrict__ qws, const f16* __restrict__ kws,
    const f16* __restrict__ vtws, float* __restrict__ out,
    const int* __restrict__ amask, int rows) {
  __shared__ __align__(32) f16 Kt[2][32 * KD];  // [key_local][dim] row-major
  __shared__ __align__(32) f16 Vt[2][KD * 32];  // [dim][key_local] (transposed)
  __shared__ __align__(32) f16 Pt[4][16 * 32];  // per-wave probability tile

  const int tid  = threadIdx.x;
  const int lane = tid & 31;
  const int wave = tid >> 5;
  const int brow0 = blockIdx.x * 64;  // block's first query row (global)
  if (brow0 >= rows) return;
  const int b   = brow0 / S_LEN;
  const int q0b = brow0 - b * S_LEN;   // block query offset within sequence
  const int q0  = q0b + wave * 16;     // this wave's query tile
  const int mask = *amask;

  const int hi = lane >> 4;
  const int n0 = lane & 15;
  const int ko = hi * 16;
  f16* pt = &Pt[wave][0];

  const f16* kbase = kws + (size_t)b * S_LEN * KD;
  const f16* vbase = vtws + (size_t)b * KD * S_LEN;

  const int kend   = mask ? (q0b + 64) : S_LEN;
  const int ntiles = kend >> 5;  // key tiles of 32 (kend is a multiple of 32)

  // Cooperative async prefetch of one 32-key tile into LDS buffer bi.
  auto prefetch = [&](int kb0, int bi) {
    // K tile: 32 rows x 64 f16 = 4KB, contiguous in workspace.
    {
      const f16* src = kbase + (size_t)kb0 * KD + tid * 16;
      unsigned   dst = lds_off(&Kt[bi][0]) + (unsigned)tid * 32;
      async_b128(dst, src);
      async_b128(dst + 16, src + 8);
    }
    // V tile: 64 dim-rows x 32 f16 (strided by S in global, packed in LDS).
    {
      const int dim = tid >> 1;
      const int hh  = (tid & 1) * 16;
      const f16* src = vbase + (size_t)dim * S_LEN + kb0 + hh;
      unsigned   dst = lds_off(&Vt[bi][0]) + (unsigned)(dim * 32 + hh) * 2;
      async_b128(dst, src);
      async_b128(dst + 16, src + 8);
    }
  };

  prefetch(0, 0);  // tile 0 in flight while we set up

  // Q A-fragments for dim chunks [0,32) and [32,64)
  const f16* qrow = qws + (size_t)(b * S_LEN + q0 + n0) * KD;
  v16h aq0, aq1;
#pragma unroll
  for (int e = 0; e < 8; ++e) {
    aq0[e] = qrow[ko + e];        aq0[e + 8] = qrow[ko + 8 + e];
    aq1[e] = qrow[32 + ko + e];   aq1[e + 8] = qrow[32 + ko + 8 + e];
  }

  v8f o[4];
#pragma unroll
  for (int nt = 0; nt < 4; ++nt) o[nt] = 0.0f;
  float mi[8], li[8];
#pragma unroll
  for (int g = 0; g < 8; ++g) { mi[g] = -__builtin_inff(); li[g] = 0.0f; }

  for (int t = 0; t < ntiles; ++t) {
    const int kb0 = t * 32;
    const int bi  = t & 1;
    // Buffer bi is in flight (own lanes) -> wait, then barrier so every
    // wave's async stores are visible and everyone is done reading bi^1.
    asm volatile("s_wait_asynccnt 0x0" ::: "memory");
    __syncthreads();
    if (t + 1 < ntiles) prefetch(kb0 + 32, bi ^ 1);

    const f16* kt = &Kt[bi][0];
    const f16* vt = &Vt[bi][0];

    // ---- scores: S = Q K^T * (1/sqrt(64)) -------------------------------
    v8f s0 = 0.0f, s1 = 0.0f;
    {
      v16h bk00 = *(const v16h*)(kt + n0 * KD + hi * 16);
      v16h bk01 = *(const v16h*)(kt + n0 * KD + 32 + hi * 16);
      v16h bk10 = *(const v16h*)(kt + (16 + n0) * KD + hi * 16);
      v16h bk11 = *(const v16h*)(kt + (16 + n0) * KD + 32 + hi * 16);
      s0 = wmma_f16(aq0, bk00, s0);
      s0 = wmma_f16(aq1, bk01, s0);
      s1 = wmma_f16(aq0, bk10, s1);
      s1 = wmma_f16(aq1, bk11, s1);
    }
    // ---- online softmax (rows spread over 16-lane halves) ---------------
    float alpha[8];
#pragma unroll
    for (int g = 0; g < 8; ++g) {
      const int m = g + 8 * hi;
      float v0 = s0[g] * 0.125f;
      float v1 = s1[g] * 0.125f;
      if (mask) {
        if (kb0 + n0 > q0 + m)      v0 = -__builtin_inff();
        if (kb0 + 16 + n0 > q0 + m) v1 = -__builtin_inff();
      }
      float r = fmaxf(v0, v1);
#pragma unroll
      for (int d = 1; d < 16; d <<= 1) r = fmaxf(r, __shfl_xor(r, d, 32));
      const float nm = fmaxf(mi[g], r);
      const float a  = __expf(mi[g] - nm);
      const float p0 = __expf(v0 - nm);
      const float p1 = __expf(v1 - nm);
      float ps = p0 + p1;
#pragma unroll
      for (int d = 1; d < 16; d <<= 1) ps += __shfl_xor(ps, d, 32);
      li[g]    = li[g] * a + ps;
      mi[g]    = nm;
      alpha[g] = a;
      pt[m * 32 + n0]      = (f16)p0;  // stage P row-major [16][32]
      pt[m * 32 + 16 + n0] = (f16)p1;
    }
#pragma unroll
    for (int nt = 0; nt < 4; ++nt) {
#pragma unroll
      for (int g = 0; g < 8; ++g) o[nt][g] *= alpha[g];
    }
    // Wave-local: DS stores of P must land before cross-lane fragment reads.
    asm volatile("s_wait_dscnt 0x0" ::: "memory");
    // ---- P A-fragment + PV WMMAs ---------------------------------------
    v16h ap;
    {
      const f16* prow = pt + n0 * 32 + ko;
#pragma unroll
      for (int e = 0; e < 8; ++e) { ap[e] = prow[e]; ap[e + 8] = prow[8 + e]; }
    }
#pragma unroll
    for (int nt = 0; nt < 4; ++nt) {
      v16h bvf = *(const v16h*)(vt + (nt * 16 + n0) * 32 + hi * 16);
      o[nt] = wmma_f16(ap, bvf, o[nt]);
    }
  }

  // ---- normalize + store fp32 ------------------------------------------
#pragma unroll
  for (int nt = 0; nt < 4; ++nt) {
#pragma unroll
    for (int g = 0; g < 8; ++g) {
      out[(size_t)(b * S_LEN + q0 + g + 8 * hi) * KD + nt * 16 + n0] =
          o[nt][g] / li[g];
    }
  }
}

// ---------------------------------------------------------------------------
extern "C" void kernel_launch(void* const* d_in, const int* in_sizes, int n_in,
                              void* d_out, int out_size, void* d_ws,
                              size_t ws_size, hipStream_t stream) {
  const float* x1 = (const float*)d_in[0];
  const float* x2 = (const float*)d_in[1];
  const float* Wq = (const float*)d_in[2];
  const float* bq = (const float*)d_in[3];
  const float* Wk = (const float*)d_in[4];
  const float* bk = (const float*)d_in[5];
  const float* Wv = (const float*)d_in[6];
  const float* bv = (const float*)d_in[7];
  const int* amask = (const int*)d_in[8];
  float* out = (float*)d_out;

  const int rows = in_sizes[0] / MD;  // B * S

  f16* qws  = (f16*)d_ws;                 // [rows][64]
  f16* kws  = qws + (size_t)rows * KD;    // [rows][64]
  f16* vtws = kws + (size_t)rows * KD;    // [B][64][S]
  f16* wT   = vtws + (size_t)rows * KD;   // [3][64][512]

  prep_weights<<<(3 * MD * KD + 255) / 256, 256, 0, stream>>>(Wq, Wk, Wv, wT);
  proj_qkv<<<rows / 64, 128, 0, stream>>>(x1, x2, wT, bq, bk, bv, qws, kws,
                                          vtws, rows);
  attn_fused<<<rows / 64, 128, 0, stream>>>(qws, kws, vtws, out, amask, rows);
}
